// LinearAttention_21208548507760
// MI455X (gfx1250) — matrix operand order repaired and checked by reading, not compiled, not run on hardware
//
#include <hip/hip_runtime.h>

#define DEVINL __device__ __forceinline__

typedef __attribute__((ext_vector_type(16))) __bf16 bf16x16;
typedef __attribute__((ext_vector_type(8)))  __bf16 bf16x8;
typedef __attribute__((ext_vector_type(8)))  float  f32x8;
typedef __attribute__((ext_vector_type(4)))  unsigned int u32x4;
typedef __attribute__((ext_vector_type(4)))  int i32x4;
typedef __attribute__((ext_vector_type(8)))  int i32x8;

static constexpr int Hn   = 16;    // heads
static constexpr int SEGc = 64;    // segment length
static constexpr int DM   = 1024;  // model dim
static constexpr int DhC  = 64;    // head dim
static constexpr int Lc   = 2048;  // sequence length
static constexpr int Rc   = Lc / SEGc; // 32 segments
static constexpr float NEGBIG = -1.0e30f;

#if defined(__has_builtin)
#if __has_builtin(__builtin_amdgcn_tensor_load_to_lds) && __has_builtin(__builtin_amdgcn_s_wait_tensorcnt)
#define HAVE_TDM 1
#endif
#endif
#ifndef HAVE_TDM
#define HAVE_TDM 0
#endif

union ABFrag { bf16x16 v; bf16x8 h[2]; };

// ---- WMMA fragment helpers (wave32; this file's contiguous K order, shared by A and B) ----
// A (16x32 bf16, MxK): lane&15 = row M; lane-half selects K 0..15 / 16..31
DEVINL bf16x16 load_a_frag_row(const __bf16* rowPtr, int k0, int lane) {
  const int hi = (lane >> 4) & 1;
  ABFrag f;
  const __bf16* p = rowPtr + k0 + hi * 16;
  f.h[0] = *reinterpret_cast<const bf16x8*>(p);
  f.h[1] = *reinterpret_cast<const bf16x8*>(p + 8);
  return f.v;
}
// B (32x16 bf16, KxN) given as Bt[N][K] (K contiguous): lane&15 = col N; half selects K 0..15 / 16..31
DEVINL bf16x16 load_b_frag(const __bf16* Bt, int ldb, int col, int k0, int lane) {
  const int hi = (lane >> 4) & 1;
  return *reinterpret_cast<const bf16x16*>(Bt + (size_t)col * ldb + k0 + hi * 16);
}
DEVINL f32x8 wmma_bf16f32(bf16x16 a, bf16x16 b, f32x8 c) {
  return __builtin_amdgcn_wmma_f32_16x16x32_bf16(false, a, false, b, (short)0, c, false, false);
}
DEVINL float group16_max(float v) {
#pragma unroll
  for (int m = 1; m <= 8; m <<= 1) v = fmaxf(v, __shfl_xor(v, m, 32));
  return v;
}
DEVINL float group16_sum(float v) {
#pragma unroll
  for (int m = 1; m <= 8; m <<= 1) v += __shfl_xor(v, m, 32);
  return v;
}

// ---------------- prep kernels ----------------
__global__ void convert_f32_to_bf16(const float* __restrict__ in, __bf16* __restrict__ out, int n) {
  int i = blockIdx.x * 256 + threadIdx.x;
  if (i < n) out[i] = (__bf16)in[i];
}
__global__ void zero_f32(float* __restrict__ p, int n) {
  int i = blockIdx.x * 256 + threadIdx.x;
  if (i < n) p[i] = 0.0f;
}

// ---------------- C[M,N] = A[M,K]bf16 @ Bt[N,K]^T, fp32 accum ----------------
// 128 threads (4 waves); block tile 128x64; each wave: 32 rows x 64 cols = 8 WMMAs/k-step.
// Software-pipelined: next k-step fragments are loaded before current WMMAs issue.
template <typename OutT>
__global__ __launch_bounds__(128)
void gemm_bf16_nt(const __bf16* __restrict__ A, const __bf16* __restrict__ Bt,
                  OutT* __restrict__ C, int M, int N, int K) {
  const int lane = threadIdx.x & 31;
  const int wave = threadIdx.x >> 5;
  const int g    = lane & 15;
  const int rowA0 = blockIdx.x * 128 + wave * 32;
  const int tileN = blockIdx.y * 64;
  const __bf16* Ar0 = A + (size_t)(rowA0 + g) * K;
  const __bf16* Ar1 = Ar0 + (size_t)16 * K;
  f32x8 acc[2][4] = {};
  bf16x16 a0 = load_a_frag_row(Ar0, 0, lane);
  bf16x16 a1 = load_a_frag_row(Ar1, 0, lane);
  bf16x16 b[4];
#pragma unroll
  for (int t = 0; t < 4; ++t) b[t] = load_b_frag(Bt, K, tileN + t * 16 + g, 0, lane);
  for (int k0 = 0; k0 < K; k0 += 32) {
    const int kn = k0 + 32;   // last iteration prefetches garbage that is never consumed
    __builtin_prefetch(Ar0 + kn + 128, 0, 0);   // global_prefetch_b8
    bf16x16 na0 = load_a_frag_row(Ar0, kn, lane);
    bf16x16 na1 = load_a_frag_row(Ar1, kn, lane);
    bf16x16 nb[4];
#pragma unroll
    for (int t = 0; t < 4; ++t) nb[t] = load_b_frag(Bt, K, tileN + t * 16 + g, kn, lane);
#pragma unroll
    for (int t = 0; t < 4; ++t) {
      acc[0][t] = wmma_bf16f32(a0, b[t], acc[0][t]);
      acc[1][t] = wmma_bf16f32(a1, b[t], acc[1][t]);
    }
    a0 = na0; a1 = na1;
#pragma unroll
    for (int t = 0; t < 4; ++t) b[t] = nb[t];
  }
#pragma unroll
  for (int rs = 0; rs < 2; ++rs) {
    const int crow = rowA0 + rs * 16 + ((lane >> 4) << 3);
#pragma unroll
    for (int t = 0; t < 4; ++t) {
      const int col = tileN + t * 16 + g;
#pragma unroll
      for (int j = 0; j < 8; ++j)
        C[(size_t)(crow + j) * N + col] = (OutT)acc[rs][t][j];
    }
  }
}

// ---------------- attention: terrace softmax + decay-weighted column sums ----------------
// grid (Rc, Hn), 128 threads (4 waves). Each block owns 64 query rows of one head.
__global__ __launch_bounds__(128)
void attn_terrace(const __bf16* __restrict__ qbf, const __bf16* __restrict__ kbf,
                  const __bf16* __restrict__ vbf, const float* __restrict__ index_decays,
                  float* __restrict__ out_t, float* __restrict__ pcs) {
  const int ri = blockIdx.x;
  const int h  = blockIdx.y;
  const int tid = threadIdx.x;
  const int lane = tid & 31, wave = tid >> 5;
  const int g = lane & 15, half = lane >> 4;

  __shared__ float sc_lds[64][130];   // terrace window scores (<=128 cols, padded)
  __shared__ float pcs_lds[Lc];
  __shared__ float mrow[64], lrow[64], ed[64];
#if HAVE_TDM
  __shared__ __bf16 vwin[128 * DhC];  // terrace-window V tile, filled by TDM
#endif

  for (int i = tid; i < Lc; i += 128) pcs_lds[i] = 0.0f;
  if (tid < 64) ed[tid] = __expf(index_decays[h * SEGc + tid]);

  const int kmax = (ri + 1) * SEGc;                 // causal key limit (exclusive)
  const int winStart = (ri > 0) ? (ri - 1) * SEGc : 0;
  const int wlen = kmax - winStart;                 // 64 or 128

#if HAVE_TDM
  if (wave == 0) {
    // Tensor Data Mover: 2D tile load of V window rows [winStart, kmax) x DhC cols into LDS.
    // D# per CDNA5 ISA 8.3-8.6: count=1, type=2(image), data_size=2B,
    // tile_dim0=DhC, tile_dim1=wlen, tensor_dim0_stride=DM.
    const unsigned long long ga =
        (unsigned long long)(const void*)(vbf + (size_t)winStart * DM + h * DhC);
    const unsigned lds_off = (unsigned)(unsigned long long)(const void*)&vwin[0];
    u32x4 g0 = {};
    g0[0] = 1u;                                          // count=1 (valid descriptor)
    g0[1] = lds_off;                                     // lds_addr
    g0[2] = (unsigned)ga;                                // global_addr[31:0]
    g0[3] = ((unsigned)(ga >> 32) & 0x01FFFFFFu) | 0x80000000u; // addr[56:32] | type=2
    i32x8 g1 = {};
    g1[0] = 0x00010000;                                  // data_size=1 -> 2 bytes
    g1[1] = (int)((DM & 0xFFFF) << 16);                  // tensor_dim0 lo16
    g1[2] = (int)((DM >> 16) & 0xFFFF) | (int)((Lc & 0xFFFF) << 16); // dim0 hi | tensor_dim1 lo
    g1[3] = (int)((Lc >> 16) & 0xFFFF) | (int)(DhC << 16);           // dim1 hi | tile_dim0
    g1[4] = wlen;                                        // tile_dim1 (tile_dim2=0)
    g1[5] = DM;                                          // tensor_dim0_stride[31:0]
    g1[6] = 0; g1[7] = 0;
    i32x4 g2 = {}, g3 = {};
#if __clang_major__ >= 23
    i32x8 g4 = {};
    __builtin_amdgcn_tensor_load_to_lds(g0, g1, g2, g3, g4, 0);
#else
    __builtin_amdgcn_tensor_load_to_lds(g0, g1, g2, g3, 0);
#endif
  }
#endif
  __syncthreads();

  const int row0 = ri * SEGc + wave * 16;
  const __bf16* Qrow = qbf + (size_t)(row0 + g) * DM + h * DhC;
  const __bf16* Kh   = kbf + h * DhC;               // Bt base, ldb = DM, col = key index

  const bf16x16 aq0 = load_a_frag_row(Qrow, 0, lane);
  const bf16x16 aq1 = load_a_frag_row(Qrow, 32, lane);

  float mj[8], lj[8];
#pragma unroll
  for (int j = 0; j < 8; ++j) { mj[j] = NEGBIG; lj[j] = 0.0f; }

  // ---- pass 1: online softmax stats (+ capture window scores), pipelined B loads ----
  {
    bf16x16 b0 = load_b_frag(Kh, DM, g, 0, lane);
    bf16x16 b1 = load_b_frag(Kh, DM, g, 32, lane);
    for (int c0 = 0; c0 < kmax; c0 += 16) {
      bf16x16 nb0 = load_b_frag(Kh, DM, c0 + 16 + g, 0, lane);
      bf16x16 nb1 = load_b_frag(Kh, DM, c0 + 16 + g, 32, lane);
      f32x8 s = {};
      s = wmma_bf16f32(aq0, b0, s);
      s = wmma_bf16f32(aq1, b1, s);
      const int col = c0 + g;
      const int rb  = row0 + half * 8;
#pragma unroll
      for (int j = 0; j < 8; ++j) {
        float v = s[j] * 0.125f;                      // 1/sqrt(64)
        if (col > rb + j) v = NEGBIG;                 // causal mask
        s[j] = v;
      }
      if (c0 >= winStart) {                           // terrace window capture
        const int lc  = col - winStart;
        const int lrb = wave * 16 + half * 8;
#pragma unroll
        for (int j = 0; j < 8; ++j) sc_lds[lrb + j][lc] = s[j];
      }
#pragma unroll
      for (int j = 0; j < 8; ++j) {
        float tmax = group16_max(s[j]);
        float mn = fmaxf(mj[j], tmax);
        float p  = __expf(s[j] - mn);
        float ps = group16_sum(p);
        lj[j] = lj[j] * __expf(mj[j] - mn) + ps;
        mj[j] = mn;
      }
      b0 = nb0; b1 = nb1;
    }
  }
  if (g == 0) {
    const int lrb = wave * 16 + half * 8;
#pragma unroll
    for (int j = 0; j < 8; ++j) { mrow[lrb + j] = mj[j]; lrow[lrb + j] = lj[j]; }
  }
#if HAVE_TDM
  if (wave == 0) __builtin_amdgcn_s_wait_tensorcnt(0);
#endif
  __syncthreads();

  // ---- terrace-window softmax + out_t (one row per thread, <=128 cols) ----
  if (tid < 64) {
    float mw = NEGBIG;
    for (int c = 0; c < wlen; ++c) mw = fmaxf(mw, sc_lds[tid][c]);
    float lw = 0.0f;
    for (int c = 0; c < wlen; ++c) {
      float p = __expf(sc_lds[tid][c] - mw);
      sc_lds[tid][c] = p;
      lw += p;
    }
    const float inv = 1.0f / lw;
    float accv[DhC];
#pragma unroll
    for (int e = 0; e < DhC; ++e) accv[e] = 0.0f;
    for (int c = 0; c < wlen; ++c) {
      const float p = sc_lds[tid][c] * inv;
#if HAVE_TDM
      const __bf16* vr = &vwin[c * DhC];
#else
      const __bf16* vr = vbf + (size_t)(winStart + c) * DM + h * DhC;
#endif
#pragma unroll
      for (int e = 0; e < DhC; ++e) accv[e] += p * (float)vr[e];
    }
    float* op = out_t + ((size_t)h * Lc + (ri * SEGc + tid)) * DhC;
    for (int e = 0; e < DhC; ++e) op[e] = accv[e];
  }

  // ---- pass 2: recompute scores, accumulate pcs[m] = sum_s e[s]*att[s,m] ----
  {
    const int lrb = wave * 16 + half * 8;
    float mloc[8], linv[8], eloc[8];
#pragma unroll
    for (int j = 0; j < 8; ++j) {
      mloc[j] = mrow[lrb + j];
      linv[j] = 1.0f / lrow[lrb + j];
      eloc[j] = ed[lrb + j];
    }
    bf16x16 b0 = load_b_frag(Kh, DM, g, 0, lane);
    bf16x16 b1 = load_b_frag(Kh, DM, g, 32, lane);
    for (int c0 = 0; c0 < kmax; c0 += 16) {
      bf16x16 nb0 = load_b_frag(Kh, DM, c0 + 16 + g, 0, lane);
      bf16x16 nb1 = load_b_frag(Kh, DM, c0 + 16 + g, 32, lane);
      f32x8 s = {};
      s = wmma_bf16f32(aq0, b0, s);
      s = wmma_bf16f32(aq1, b1, s);
      const int col = c0 + g;
      const int rb  = row0 + half * 8;
      float cs = 0.0f;
#pragma unroll
      for (int j = 0; j < 8; ++j) {
        float v = s[j] * 0.125f;
        if (col > rb + j) v = NEGBIG;
        cs += eloc[j] * __expf(v - mloc[j]) * linv[j];
      }
      cs += __shfl_xor(cs, 16, 32);                 // combine row halves
      if (half == 0) atomicAdd(&pcs_lds[col], cs);  // ds_add_f32
      b0 = nb0; b1 = nb1;
    }
  }
  __syncthreads();
  float* pp = pcs + ((size_t)h * Rc + ri) * Lc;
  for (int i = tid; i < Lc; i += 128) pp[i] = pcs_lds[i];
}

// ---------------- mega-decay scan over segments + row sums ----------------
__global__ __launch_bounds__(256)
void scan_kernel(const float* __restrict__ pcs, const float* __restrict__ mega_decays,
                 float* __restrict__ w, float* __restrict__ wsum) {
  const int h = blockIdx.y;
  const int m = blockIdx.x * 256 + threadIdx.x;
  float mg = __expf(mega_decays[h] * (float)SEGc);
  mg = fminf(fmaxf(mg, 0.0f), 2.0f);
  __shared__ float part[Rc];
  if (threadIdx.x < Rc) part[threadIdx.x] = 0.0f;
  __syncthreads();
  float carry = 0.0f;
  for (int i = 0; i < Rc; ++i) {
    const size_t off = ((size_t)h * Rc + i) * Lc + m;
    const float outv = carry;
    carry = carry * mg + pcs[off];
    w[off] = outv;
    atomicAdd(&part[i], outv);
  }
  __syncthreads();
  if (threadIdx.x < Rc) atomicAdd(&wsum[h * Rc + threadIdx.x], part[threadIdx.x]);
}

// ---------------- kv[r,h] = sum_s wn[s] * k[s]^T v[s]  (64x64xK WMMA GEMM) ----------------
// grid (Rc, Hn), 256 threads (8 waves). Double-buffered LDS staging. Stores kv^T: kvT[e][d].
__global__ __launch_bounds__(256)
void kv_kernel(const __bf16* __restrict__ kbf, const __bf16* __restrict__ vbf,
               const float* __restrict__ w, const float* __restrict__ wsum,
               __bf16* __restrict__ kvT) {
  const int ri = blockIdx.x, h = blockIdx.y;
  const int tid = threadIdx.x, lane = tid & 31, wave = tid >> 5;
  const int g = lane & 15;
  __shared__ __bf16 Alds[2][64][32];   // [buf][d][s] = k^T scaled by wn
  __shared__ __bf16 Blds[2][64][32];   // [buf][e][s] = v^T
  const int dblk = wave & 3, eg = wave >> 2;
  f32x8 acc[2] = {};
  const float inv = 1.0f / (wsum[h * Rc + ri] + 1e-5f);
  const float* wrow = w + ((size_t)h * Rc + ri) * Lc;
  const int nT = (ri * SEGc) / 32;     // w[...] == 0 for s >= ri*64

  auto stage = [&](int buf, int s0) {
    for (int idx = tid; idx < 64 * 32; idx += 256) {
      const int d = idx & 63, s = idx >> 6;
      const float wn = wrow[s0 + s] * inv;
      Alds[buf][d][s] = (__bf16)((float)kbf[(size_t)(s0 + s) * DM + h * DhC + d] * wn);
      Blds[buf][d][s] = vbf[(size_t)(s0 + s) * DM + h * DhC + d];
    }
  };

  if (nT > 0) stage(0, 0);
  __syncthreads();
  for (int i = 0; i < nT; ++i) {
    const int cur = i & 1;
    if (i + 1 < nT) stage(1 - cur, (i + 1) * 32);   // overlap staging with WMMA
    bf16x16 af = load_a_frag_row(&Alds[cur][dblk * 16 + g][0], 0, lane);
#pragma unroll
    for (int t = 0; t < 2; ++t) {
      bf16x16 bfr = load_b_frag(&Blds[cur][0][0], 32, (eg * 2 + t) * 16 + g, 0, lane);
      acc[t] = wmma_bf16f32(af, bfr, acc[t]);
    }
    __syncthreads();
  }
  const int dr = dblk * 16 + (lane >> 4) * 8;
  __bf16* base = kvT + ((size_t)ri * Hn + h) * (DhC * DhC);
#pragma unroll
  for (int t = 0; t < 2; ++t) {
    const int e = (eg * 2 + t) * 16 + g;
#pragma unroll
    for (int j = 0; j < 8; ++j)
      base[(size_t)e * DhC + (dr + j)] = (__bf16)acc[t][j];
  }
}

// ---------------- out_lin = q @ kv; y = tw*out_t + (1-tw)*out_lin ----------------
// grid (Rc, Hn), 128 threads (4 waves).
__global__ __launch_bounds__(128)
void mixo_kernel(const __bf16* __restrict__ qbf, const __bf16* __restrict__ kvT,
                 const float* __restrict__ out_t, const float* __restrict__ terrace_mix,
                 __bf16* __restrict__ ybf) {
  const int ri = blockIdx.x, h = blockIdx.y;
  const int lane = threadIdx.x & 31, wave = threadIdx.x >> 5;
  const int g = lane & 15;
  const int grow = ri * SEGc + wave * 16 + g;
  const __bf16* Qrow = qbf + (size_t)grow * DM + h * DhC;
  const __bf16* KV   = kvT + ((size_t)ri * Hn + h) * (DhC * DhC);
  f32x8 acc[4] = {};
#pragma unroll
  for (int k0 = 0; k0 < DhC; k0 += 32) {
    bf16x16 af = load_a_frag_row(Qrow, k0, lane);
#pragma unroll
    for (int t = 0; t < 4; ++t) {
      bf16x16 bfr = load_b_frag(KV, DhC, t * 16 + g, k0, lane);
      acc[t] = wmma_bf16f32(af, bfr, acc[t]);
    }
  }
  const int lr0 = wave * 16 + (lane >> 4) * 8;
#pragma unroll
  for (int t = 0; t < 4; ++t) {
    const int e = t * 16 + g;
#pragma unroll
    for (int j = 0; j < 8; ++j) {
      const int lr = lr0 + j;
      const int row = ri * SEGc + lr;
      const float tw = 1.0f / (1.0f + __expf(-terrace_mix[lr]));
      const float ot = out_t[((size_t)h * Lc + row) * DhC + e];
      const float yv = tw * ot + (1.0f - tw) * acc[t][j];
      ybf[(size_t)row * DM + h * DhC + e] = (__bf16)yv;
    }
  }
}

// ---------------- host launch ----------------
extern "C" void kernel_launch(void* const* d_in, const int* in_sizes, int n_in,
                              void* d_out, int out_size, void* d_ws, size_t ws_size,
                              hipStream_t stream) {
  (void)in_sizes; (void)n_in; (void)out_size; (void)ws_size;
  const float* hs           = (const float*)d_in[0];
  const float* Wq           = (const float*)d_in[1];
  const float* Wk           = (const float*)d_in[2];
  const float* Wv           = (const float*)d_in[3];
  const float* Wc           = (const float*)d_in[4];
  const float* terrace_mix  = (const float*)d_in[5];
  const float* index_decays = (const float*)d_in[6];
  const float* mega_decays  = (const float*)d_in[7];
  float* out = (float*)d_out;

  char* ws = (char*)d_ws;
  size_t off = 0;
  auto wsalloc = [&](size_t bytes) -> void* {
    void* p = ws + off;
    off += (bytes + 255) & ~(size_t)255;
    return p;
  };
  __bf16* hs_bf = (__bf16*)wsalloc((size_t)Lc * DM * 2);
  __bf16* wq_bf = (__bf16*)wsalloc((size_t)DM * DM * 2);
  __bf16* wk_bf = (__bf16*)wsalloc((size_t)DM * DM * 2);
  __bf16* wv_bf = (__bf16*)wsalloc((size_t)DM * DM * 2);
  __bf16* wc_bf = (__bf16*)wsalloc((size_t)DM * DM * 2);
  __bf16* q_bf  = (__bf16*)wsalloc((size_t)Lc * DM * 2);
  __bf16* k_bf  = (__bf16*)wsalloc((size_t)Lc * DM * 2);
  __bf16* v_bf  = (__bf16*)wsalloc((size_t)Lc * DM * 2);
  __bf16* y_bf  = (__bf16*)wsalloc((size_t)Lc * DM * 2);
  float*  pcs   = (float*)wsalloc((size_t)Hn * Rc * Lc * 4);
  float*  wbuf  = (float*)wsalloc((size_t)Hn * Rc * Lc * 4);
  float*  wsum  = (float*)wsalloc((size_t)Hn * Rc * 4);
  float*  out_t = (float*)wsalloc((size_t)Hn * Lc * DhC * 4);
  __bf16* kvT   = (__bf16*)wsalloc((size_t)Rc * Hn * DhC * DhC * 2);

  // 0) convert to bf16, zero accumulators
  const int nHS = Lc * DM, nW = DM * DM;
  convert_f32_to_bf16<<<(nHS + 255) / 256, 256, 0, stream>>>(hs, hs_bf, nHS);
  convert_f32_to_bf16<<<(nW + 255) / 256, 256, 0, stream>>>(Wq, wq_bf, nW);
  convert_f32_to_bf16<<<(nW + 255) / 256, 256, 0, stream>>>(Wk, wk_bf, nW);
  convert_f32_to_bf16<<<(nW + 255) / 256, 256, 0, stream>>>(Wv, wv_bf, nW);
  convert_f32_to_bf16<<<(nW + 255) / 256, 256, 0, stream>>>(Wc, wc_bf, nW);
  zero_f32<<<(Hn * Rc + 255) / 256, 256, 0, stream>>>(wsum, Hn * Rc);

  // 1) Q/K/V projections: (2048x1024) @ (1024x1024)^T, bf16 WMMA
  dim3 gGemm(Lc / 128, DM / 64);
  gemm_bf16_nt<__bf16><<<gGemm, 128, 0, stream>>>(hs_bf, wq_bf, q_bf, Lc, DM, DM);
  gemm_bf16_nt<__bf16><<<gGemm, 128, 0, stream>>>(hs_bf, wk_bf, k_bf, Lc, DM, DM);
  gemm_bf16_nt<__bf16><<<gGemm, 128, 0, stream>>>(hs_bf, wv_bf, v_bf, Lc, DM, DM);

  // 2) attention: terrace softmax output + decay-weighted per-segment column sums
  attn_terrace<<<dim3(Rc, Hn), 128, 0, stream>>>(q_bf, k_bf, v_bf, index_decays, out_t, pcs);

  // 3) mega-decay scan over segments -> w, row sums
  scan_kernel<<<dim3(Lc / 256, Hn), 256, 0, stream>>>(pcs, mega_decays, wbuf, wsum);

  // 4) kv state per (segment, head), stored transposed
  kv_kernel<<<dim3(Rc, Hn), 256, 0, stream>>>(k_bf, v_bf, wbuf, wsum, kvT);

  // 5) out_lin = q @ kv, sigmoid mix with terrace output -> y (bf16)
  mixo_kernel<<<dim3(Rc, Hn), 128, 0, stream>>>(q_bf, kvT, out_t, terrace_mix, y_bf);

  // 6) final projection y @ Wc^T -> fp32 output
  gemm_bf16_nt<float><<<gGemm, 128, 0, stream>>>(y_bf, wc_bf, out, Lc, DM, DM);
}
